// MultiHeadAttention_26946624815928
// MI455X (gfx1250) — matrix operand order repaired and checked by reading, not compiled, not run on hardware
//
#include <hip/hip_runtime.h>
#include <hip/hip_bf16.h>

typedef __attribute__((ext_vector_type(16))) _Float16 v16h;
typedef __attribute__((ext_vector_type(8)))  _Float16 v8h;
typedef __attribute__((ext_vector_type(8)))  float    v8f;

#define BATCH  2
#define SEQ    2048
#define DMODEL 1024
#define HEADS  16
#define HDIM   64
#define MROWS  (BATCH * SEQ)   // 4096
#define ATTN_SCALE 0.125f      // 1/sqrt(64), exact power of two

__device__ __forceinline__ v8f wmma_f16(v16h a, v16h b, v8f c) {
  return __builtin_amdgcn_wmma_f32_16x16x32_f16(
      /*neg_a=*/false, a, /*neg_b=*/false, b,
      /*c_mod=*/(short)0, c, /*reuse_a=*/false, /*reuse_b=*/false);
}

// 16 contiguous f16 (32B) -> B-fragment (two b128 loads)
__device__ __forceinline__ v16h ld16h(const _Float16* p) {
  union { v16h v; v8h h[2]; } u;
  u.h[0] = *(const v8h*)(p);
  u.h[1] = *(const v8h*)(p + 8);
  return u.v;
}

// A-fragment (16x32 f16, MxK): lane m=lane&15, halves = K[kb..kb+16) contiguous
__device__ __forceinline__ v16h frag_a_f16(const _Float16* row, int lane) {
  const int kb = (lane & 16) ? 16 : 0;
  union { v16h v; v8h h[2]; } u;
  u.h[0] = *(const v8h*)(row + kb);
  u.h[1] = *(const v8h*)(row + kb + 8);
  return u.v;
}

// Same A-fragment, converting from f32 source in-register
__device__ __forceinline__ v16h frag_a_f32(const float* row, int lane) {
  const int kb = (lane & 16) ? 16 : 0;
  v16h r;
#pragma unroll
  for (int i = 0; i < 8; ++i) {
    r[i]     = (_Float16)row[kb + i];
    r[i + 8] = (_Float16)row[kb + 8 + i];
  }
  return r;
}

// ---- CDNA5 async global->LDS copy (ASYNCcnt), inline asm per ISA 10/15.18 ----
__device__ __forceinline__ void async_ld_b128(unsigned lds_off, const _Float16* g) {
  asm volatile("global_load_async_to_lds_b128 %0, %1, off"
               :: "v"(lds_off), "v"(g) : "memory");
}
__device__ __forceinline__ void wait_async_le2() {
  asm volatile("s_wait_asynccnt 0x2" ::: "memory");
}

// ---------------------------------------------------------------------------
// Weight pack: W (f32, [K][N]) -> Wt (f16, [N][K]).
// ---------------------------------------------------------------------------
__global__ void __launch_bounds__(256)
pack_wt(const float* __restrict__ W, _Float16* __restrict__ Wt) {
  int idx = blockIdx.x * 256 + threadIdx.x;  // 1M elements
  int n = idx >> 10;
  int k = idx & (DMODEL - 1);
  Wt[idx] = (_Float16)W[(size_t)k * DMODEL + n];
}

// ---------------------------------------------------------------------------
// GEMM + bias, software-pipelined 32x64 per wave, K-step 32.
// mode 0: f16 head-major out[((b*H+h)*S + s)*64 + hd]              (k)
// mode 1: f16 head-major, value scaled by ATTN_SCALE               (q)
// mode 2: f16 head-transposed out[((b*H+h)*64 + hd)*S + s]         (v)
// mode 3: f32 out[m*1024 + n]                                      (final)
// ---------------------------------------------------------------------------
template <bool AF16>
__global__ void __launch_bounds__(256)
gemm_bias_t(const void* __restrict__ Av, const _Float16* __restrict__ Bt,
            const float* __restrict__ bias, void* __restrict__ outv, int mode) {
  const int lane = threadIdx.x & 31;
  const int widx = threadIdx.x >> 5;
  const int wid  = blockIdx.x * 8 + widx;      // 0..2047
  const int m0   = (wid >> 4) * 32;
  const int n0   = (wid & 15) * 64;
  const int lm   = lane & 15;
  const int hi   = (lane & 16) ? 1 : 0;

  const float*    A32 = (const float*)Av;
  const _Float16* A16 = (const _Float16*)Av;
  const _Float16* Brow = Bt + (size_t)(n0 + lm) * DMODEL + (hi ? 16 : 0);

  const v8f zero8 = {0.f, 0.f, 0.f, 0.f, 0.f, 0.f, 0.f, 0.f};
  v8f acc[2][4];
#pragma unroll
  for (int i = 0; i < 2; ++i)
#pragma unroll
    for (int j = 0; j < 4; ++j) acc[i][j] = zero8;

  auto loadA = [&](int k, v16h& a0, v16h& a1) {
    if (AF16) {
      a0 = frag_a_f16(A16 + (size_t)(m0 + lm)      * DMODEL + k, lane);
      a1 = frag_a_f16(A16 + (size_t)(m0 + 16 + lm) * DMODEL + k, lane);
    } else {
      a0 = frag_a_f32(A32 + (size_t)(m0 + lm)      * DMODEL + k, lane);
      a1 = frag_a_f32(A32 + (size_t)(m0 + 16 + lm) * DMODEL + k, lane);
    }
  };

  // ---- prologue: preload k = 0 ----
  v16h a0, a1, b[4];
  loadA(0, a0, a1);
#pragma unroll
  for (int nt = 0; nt < 4; ++nt) b[nt] = ld16h(Brow + (size_t)nt * 16 * DMODEL);

  // ---- pipelined main loop ----
  for (int k = 32; k < DMODEL; k += 32) {
    v16h na0, na1, nb[4];
    loadA(k, na0, na1);
#pragma unroll
    for (int nt = 0; nt < 4; ++nt)
      nb[nt] = ld16h(Brow + (size_t)nt * 16 * DMODEL + k);

    if (AF16) {
      __builtin_prefetch(A16 + (size_t)(m0 + lm) * DMODEL + k + 32, 0, 3);
    } else {
      __builtin_prefetch(A32 + (size_t)(m0 + lm) * DMODEL + k + 32, 0, 3);
    }

#pragma unroll
    for (int nt = 0; nt < 4; ++nt) {
      acc[0][nt] = wmma_f16(a0, b[nt], acc[0][nt]);
      acc[1][nt] = wmma_f16(a1, b[nt], acc[1][nt]);
    }
    a0 = na0; a1 = na1;
#pragma unroll
    for (int nt = 0; nt < 4; ++nt) b[nt] = nb[nt];
  }

  // ---- drain ----
#pragma unroll
  for (int nt = 0; nt < 4; ++nt) {
    acc[0][nt] = wmma_f16(a0, b[nt], acc[0][nt]);
    acc[1][nt] = wmma_f16(a1, b[nt], acc[1][nt]);
  }

  // ---- epilogue ----
#pragma unroll
  for (int mt = 0; mt < 2; ++mt) {
#pragma unroll
    for (int nt = 0; nt < 4; ++nt) {
#pragma unroll
      for (int r = 0; r < 8; ++r) {
        int m = m0 + mt * 16 + r + hi * 8;
        int n = n0 + nt * 16 + lm;
        float val = acc[mt][nt][r] + bias[n];
        if (mode == 3) {
          ((float*)outv)[(size_t)m * DMODEL + n] = val;
        } else {
          if (mode == 1) val *= ATTN_SCALE;
          int bb = m >> 11;            // / SEQ
          int s  = m & (SEQ - 1);
          int h  = n >> 6;
          int hd = n & 63;
          int bh = bb * HEADS + h;
          size_t idx = (mode == 2)
              ? ((size_t)bh * HDIM + hd) * SEQ + s
              : ((size_t)bh * SEQ + s) * HDIM + hd;
          ((_Float16*)outv)[idx] = (_Float16)val;
        }
      }
    }
  }
}

// ---------------------------------------------------------------------------
// Fused attention, transposed dataflow + block-cooperative async staging:
// all 8 waves of a block share one (batch,head); the K chunk (32x64) and
// V^T chunk (64x32) are staged into LDS once per block via
// global_load_async_to_lds_b128, double-buffered, and consumed as WMMA
// fragments with ds_load_b128.
//   S^T = K * Q^T   (K rows from LDS = A-fragments, Q rows = B-fragments)
//   O^T = V^T * P^T (vT rows from LDS = A-fragments, P^T from LDS = B-frag)
// qh (pre-scaled)/kh: [BH][S][64] f16; vT: [BH][64][S] f16.
// ---------------------------------------------------------------------------
__global__ void __launch_bounds__(256)
attn_fused(const _Float16* __restrict__ qh, const _Float16* __restrict__ kh,
           const _Float16* __restrict__ vT, _Float16* __restrict__ ctx) {
  __shared__ __attribute__((aligned(16))) _Float16 kstage[2][32 * HDIM]; // 8 KB
  __shared__ __attribute__((aligned(16))) _Float16 vstage[2][HDIM * 32]; // 8 KB
  __shared__ __attribute__((aligned(16))) _Float16 pbuf[8][16 * 32];     // 8 KB

  const int tid  = threadIdx.x;             // 0..255
  const int lane = tid & 31;
  const int widx = tid >> 5;
  const int bh   = blockIdx.x >> 4;                        // shared by block
  const int q0   = (((blockIdx.x & 15) << 3) + widx) * 16; // per-wave q tile
  const int bb   = bh >> 4;
  const int h    = bh & 15;
  const int lm   = lane & 15;
  const int hi   = (lane & 16) ? 1 : 0;
  const int hioff = hi ? 16 : 0;

  // Q as B-operand: lane holds query column q0+lm, contiguous hd halves
  const _Float16* qrow = qh + ((size_t)bh * SEQ + q0 + lm) * HDIM + hioff;
  v16h bq0 = ld16h(qrow);        // hd step 0 (0..31 across lane halves)
  v16h bq1 = ld16h(qrow + 32);   // hd step 1 (32..63)

  // staging source pointers (per thread, 8 halves = one b128 each)
  const _Float16* ksrc = kh + (size_t)bh * SEQ * HDIM + tid * 8;         // + kv*64
  const int vr = tid >> 2, vc = (tid & 3) * 8;
  const _Float16* vsrc = vT + ((size_t)bh * HDIM + vr) * SEQ + vc;       // + kv

  auto stage = [&](int kv_, int buf) {
    async_ld_b128((unsigned)(size_t)&kstage[buf][tid * 8], ksrc + (size_t)kv_ * HDIM);
    async_ld_b128((unsigned)(size_t)&vstage[buf][vr * 32 + vc], vsrc + kv_);
  };

  const v8f zero8 = {0.f, 0.f, 0.f, 0.f, 0.f, 0.f, 0.f, 0.f};
  v8f o[4];                 // O^T: M=hd on VGPRs/half, N=query on lanes
#pragma unroll
  for (int i = 0; i < 4; ++i) o[i] = zero8;
  float rm = -1e30f, rs = 0.f;   // per-lane (per-query) running max / sum

  stage(0, 0);  // prologue fill of buffer 0 (2 async ops outstanding)

  for (int kv = 0; kv < SEQ; kv += 32) {
    const int c   = kv >> 5;
    const int cur = c & 1;
    const int nxt = cur ^ 1;

    // issue fills for next chunk, then wait for current chunk's fills
    // (in-order async completion: <=2 outstanding means the older 2 are done)
    stage((kv + 32) & (SEQ - 1), nxt);
    wait_async_le2();
    __syncthreads();   // everyone's fills for `cur` are complete

    // ---- S^T tiles from LDS-staged K: s0 = keys[0..16), s1 = keys[16..32) ----
    v8f s0 = zero8, s1 = zero8;
    s0 = wmma_f16(frag_a_f16(&kstage[cur][(lm)      * HDIM],      lane), bq0, s0);
    s0 = wmma_f16(frag_a_f16(&kstage[cur][(lm)      * HDIM + 32], lane), bq1, s0);
    s1 = wmma_f16(frag_a_f16(&kstage[cur][(16 + lm) * HDIM],      lane), bq0, s1);
    s1 = wmma_f16(frag_a_f16(&kstage[cur][(16 + lm) * HDIM + 32], lane), bq1, s1);

    // V fragments from LDS (used after softmax)
    v16h av[4];
#pragma unroll
    for (int nt = 0; nt < 4; ++nt)
      av[nt] = frag_a_f16(&vstage[cur][(nt * 16 + lm) * 32], lane);

    // ---- online softmax: per-lane tree + one xor-16 shuffle ----
    float mx = fmaxf(s0[0], s1[0]);
#pragma unroll
    for (int r = 1; r < 8; ++r) mx = fmaxf(mx, fmaxf(s0[r], s1[r]));
    mx = fmaxf(mx, __shfl_xor(mx, 16, 32));
    float mn   = fmaxf(rm, mx);
    float corr = __expf(rm - mn);
    rm = mn;

    float e0[8], e1[8], ps = 0.f;
#pragma unroll
    for (int r = 0; r < 8; ++r) {
      e0[r] = __expf(s0[r] - mn);
      e1[r] = __expf(s1[r] - mn);
      ps += e0[r] + e1[r];
    }
    ps += __shfl_xor(ps, 16, 32);
    rs = rs * corr + ps;
#pragma unroll
    for (int nt = 0; nt < 4; ++nt)
#pragma unroll
      for (int r = 0; r < 8; ++r) o[nt][r] *= corr;

    // ---- pack P^T into LDS: [query][key], 8 consecutive keys per half ----
    v8h pk0, pk1;
#pragma unroll
    for (int r = 0; r < 8; ++r) { pk0[r] = (_Float16)e0[r]; pk1[r] = (_Float16)e1[r]; }
    *(v8h*)&pbuf[widx][lm * 32 + hi * 8]      = pk0;   // keys t0: r + 8*hi
    *(v8h*)&pbuf[widx][lm * 32 + 16 + hi * 8] = pk1;   // keys t1

    __syncthreads();  // uniform trip count across all 8 waves

    // ---- O^T += V^T * P^T : P^T is the B-fragment straight from LDS ----
    v16h bp = ld16h(&pbuf[widx][lm * 32 + hioff]);
#pragma unroll
    for (int nt = 0; nt < 4; ++nt)
      o[nt] = wmma_f16(av[nt], bp, o[nt]);

    __syncthreads();  // all reads of `cur` done before it is refilled next iter
  }

  // ---- normalize; packed b128 stores (8 consecutive hd per half-lane) ----
  const float inv = 1.0f / rs;
  _Float16* crow =
      ctx + ((size_t)bb * SEQ + q0 + lm) * DMODEL + h * HDIM + hi * 8;
#pragma unroll
  for (int nt = 0; nt < 4; ++nt) {
    v8h pk;
#pragma unroll
    for (int r = 0; r < 8; ++r) pk[r] = (_Float16)(o[nt][r] * inv);
    *(v8h*)(crow + nt * 16) = pk;
  }
}

// ---------------------------------------------------------------------------
extern "C" void kernel_launch(void* const* d_in, const int* in_sizes, int n_in,
                              void* d_out, int out_size, void* d_ws, size_t ws_size,
                              hipStream_t stream) {
  (void)in_sizes; (void)n_in; (void)out_size; (void)ws_size;
  const float* Q  = (const float*)d_in[0];
  const float* K  = (const float*)d_in[1];
  const float* V  = (const float*)d_in[2];
  const float* Wq = (const float*)d_in[3];
  const float* bq = (const float*)d_in[4];
  const float* Wk = (const float*)d_in[5];
  const float* bk = (const float*)d_in[6];
  const float* Wv = (const float*)d_in[7];
  const float* bv = (const float*)d_in[8];
  const float* Wo = (const float*)d_in[9];
  const float* bo = (const float*)d_in[10];

  char* ws = (char*)d_ws;
  const size_t WT  = (size_t)DMODEL * DMODEL * sizeof(_Float16);  // 2 MB
  const size_t ACT = (size_t)MROWS * DMODEL * sizeof(_Float16);   // 8 MB
  _Float16* wtq = (_Float16*)(ws);
  _Float16* wtk = (_Float16*)(ws + WT);
  _Float16* wtv = (_Float16*)(ws + 2 * WT);
  _Float16* wto = (_Float16*)(ws + 3 * WT);
  _Float16* qws = (_Float16*)(ws + 4 * WT);            // [BH][S][64], pre-scaled
  _Float16* kws = (_Float16*)(ws + 4 * WT + ACT);      // [BH][S][64]
  _Float16* vws = (_Float16*)(ws + 4 * WT + 2 * ACT);  // [BH][64][S]
  _Float16* ctx = (_Float16*)(ws + 4 * WT + 3 * ACT);  // [B*S][1024]

  dim3 blk(256);
  pack_wt<<<4096, blk, 0, stream>>>(Wq, wtq);
  pack_wt<<<4096, blk, 0, stream>>>(Wk, wtk);
  pack_wt<<<4096, blk, 0, stream>>>(Wv, wtv);
  pack_wt<<<4096, blk, 0, stream>>>(Wo, wto);

  gemm_bias_t<false><<<256, blk, 0, stream>>>((const void*)Q, wtq, bq, (void*)qws, 1);
  gemm_bias_t<false><<<256, blk, 0, stream>>>((const void*)K, wtk, bk, (void*)kws, 0);
  gemm_bias_t<false><<<256, blk, 0, stream>>>((const void*)V, wtv, bv, (void*)vws, 2);

  attn_fused<<<512, blk, 0, stream>>>(qws, kws, vws, ctx);

  gemm_bias_t<true><<<256, blk, 0, stream>>>((const void*)ctx, wto, bo, d_out, 3);
}
